// Model_24592982737624
// MI455X (gfx1250) — compile-verified, matrix operand order checked by reading, not run on hardware
//
#include <hip/hip_runtime.h>
#include <hip/hip_bf16.h>
#include <math.h>

// Problem constants (from the reference)
constexpr int B    = 4;
constexpr int N    = 10000;
constexpr int E    = 163840;
constexpr int NREG = 2000;
constexpr int DIN  = 64;
constexpr int DEN  = 16;
constexpr float NEG_SLOPE = 0.2f;

typedef __attribute__((ext_vector_type(16))) _Float16 v16h;
typedef __attribute__((ext_vector_type(8)))  float    v8f;

// ---------------------------------------------------------------------------
// float atomic max via int/uint ordering trick (correct for all sign mixes)
// ---------------------------------------------------------------------------
__device__ __forceinline__ void atomicMaxF(float* addr, float val) {
    if (val >= 0.0f) {
        atomicMax((int*)addr, __float_as_int(val));
    } else {
        atomicMin((unsigned int*)addr, __float_as_uint(val));
    }
}

// ---------------------------------------------------------------------------
// K0: init workspace (m = -inf, den/rep/pool1 = 0) + edge coefficient
// ---------------------------------------------------------------------------
__global__ void k_init(float* __restrict__ m, float* __restrict__ den,
                       float* __restrict__ rep, float* __restrict__ pool1,
                       float* __restrict__ ce,
                       const float* __restrict__ lin_edge,
                       const float* __restrict__ att_edge) {
    int idx = blockIdx.x * blockDim.x + threadIdx.x;
    if (idx == 0) {
        float s = 0.0f;
        #pragma unroll
        for (int c = 0; c < DEN; ++c) s += lin_edge[c] * att_edge[c];
        *ce = s;
    }
    if (idx < B * N) { m[idx] = -INFINITY; den[idx] = 0.0f; }
    if (idx < B * N * DEN)    rep[idx]   = 0.0f;
    if (idx < B * NREG * DEN) pool1[idx] = 0.0f;
}

// ---------------------------------------------------------------------------
// K1: h = x @ W via v_wmma_f32_16x16x32_f16  (one wave per 16-row tile)
// A layout: lane L holds row L%16; halves j: j<8 -> K = kt*32 + (L/16)*8 + j
//                                         j>=8 -> K = kt*32 + 16 + (L/16)*8 + (j-8)
// B layout mirrors A with column L%16 of W.
// C/D layout: c[r] @ lane L -> row r + 8*(L/16), col L%16.
// ---------------------------------------------------------------------------
__global__ void k_gemm_wmma(const float* __restrict__ x,
                            const float* __restrict__ Wm,
                            float* __restrict__ h) {
    const int wavesPerBlock = blockDim.x >> 5;
    const int wave = blockIdx.x * wavesPerBlock + (threadIdx.x >> 5);
    const int lane = threadIdx.x & 31;
    constexpr int TILES = (B * N) / 16;   // 2500, exact
    if (wave >= TILES) return;            // wave-uniform: EXEC stays all-ones for WMMA

    const int row0 = wave * 16;
    const int mrow = lane & 15;           // row within tile for A; column for B/C/D
    const int grp  = lane >> 4;
    const float* xr = x + (long long)(row0 + mrow) * DIN;

    v8f c = {};
    #pragma unroll
    for (int kt = 0; kt < 2; ++kt) {
        const int kbase = kt * 32;
        v16h a, b;
        #pragma unroll
        for (int j = 0; j < 8; ++j) {
            const int k0 = kbase + grp * 8 + j;
            const int k1 = kbase + 16 + grp * 8 + j;
            a[j]     = (_Float16)xr[k0];
            a[j + 8] = (_Float16)xr[k1];
            b[j]     = (_Float16)Wm[k0 * DEN + mrow];
            b[j + 8] = (_Float16)Wm[k1 * DEN + mrow];
        }
        c = __builtin_amdgcn_wmma_f32_16x16x32_f16(
                /*neg_a=*/false, a, /*neg_b=*/false, b,
                /*c_mod=*/(short)0, c, /*reuse_a=*/false, /*reuse_b=*/false);
    }

    float* hr = h + (long long)row0 * DEN;
    #pragma unroll
    for (int r = 0; r < 8; ++r) {
        const int rr = r + 8 * grp;
        hr[rr * DEN + mrow] = c[r];
    }
}

// ---------------------------------------------------------------------------
// K2: per-node attention projections a_s = h.att_src, a_d = h.att_dst
// ---------------------------------------------------------------------------
__global__ void k_node_proj(const float* __restrict__ h,
                            const float* __restrict__ att_src,
                            const float* __restrict__ att_dst,
                            float* __restrict__ a_s, float* __restrict__ a_d) {
    int idx = blockIdx.x * blockDim.x + threadIdx.x;
    if (idx >= B * N) return;
    const float* hr = h + (long long)idx * DEN;
    float s = 0.0f, d = 0.0f;
    #pragma unroll
    for (int c = 0; c < DEN; ++c) { s += hr[c] * att_src[c]; d += hr[c] * att_dst[c]; }
    a_s[idx] = s; a_d[idx] = d;
}

// ---------------------------------------------------------------------------
// K3: per-edge alpha = leakyrelu(a_s[src]+a_d[dst]+ce*ea), atomic segment max
// ---------------------------------------------------------------------------
__global__ void k_alpha_max(const long long* __restrict__ ei,
                            const float* __restrict__ ea,
                            const float* __restrict__ a_s,
                            const float* __restrict__ a_d,
                            const float* __restrict__ ce,
                            float* __restrict__ alpha,
                            float* __restrict__ m) {
    int idx = blockIdx.x * blockDim.x + threadIdx.x;
    if (idx >= B * E) return;
    const int b = idx / E, e = idx - b * E;
    const long long* src = ei + (long long)b * 2 * E;
    const long long* dst = src + E;
    const int s = (int)src[e], d = (int)dst[e];
    float al = a_s[b * N + s] + a_d[b * N + d] + ea[idx] * (*ce);
    al = (al > 0.0f) ? al : NEG_SLOPE * al;
    alpha[idx] = al;
    atomicMaxF(&m[b * N + d], al);
}

// ---------------------------------------------------------------------------
// K4: ex = exp(alpha - m[dst]) (in place), atomic segment sum into den
// ---------------------------------------------------------------------------
__global__ void k_exp_sum(const long long* __restrict__ ei,
                          const float* __restrict__ m,
                          float* __restrict__ alpha,  // becomes ex
                          float* __restrict__ den) {
    int idx = blockIdx.x * blockDim.x + threadIdx.x;
    if (idx >= B * E) return;
    const int b = idx / E, e = idx - b * E;
    const long long* dst = ei + (long long)b * 2 * E + E;
    const int d = (int)dst[e];
    const float ex = __expf(alpha[idx] - m[b * N + d]);
    alpha[idx] = ex;
    atomicAdd(&den[b * N + d], ex);
}

// ---------------------------------------------------------------------------
// K5: rep[dst] += (ex/den[dst]) * h[src]   (16 float atomics per edge)
// ---------------------------------------------------------------------------
__global__ void k_scatter(const long long* __restrict__ ei,
                          const float* __restrict__ ex,
                          const float* __restrict__ den,
                          const float* __restrict__ h,
                          float* __restrict__ rep) {
    int idx = blockIdx.x * blockDim.x + threadIdx.x;
    if (idx >= B * E) return;
    const int b = idx / E, e = idx - b * E;
    const long long* src = ei + (long long)b * 2 * E;
    const long long* dst = src + E;
    const int s = (int)src[e], d = (int)dst[e];
    const float attn = ex[idx] / den[b * N + d];
    const float* hs = h + (long long)(b * N + s) * DEN;
    float* rd = rep + (long long)(b * N + d) * DEN;
    #pragma unroll
    for (int c = 0; c < DEN; ++c) atomicAdd(&rd[c], attn * hs[c]);
}

// ---------------------------------------------------------------------------
// K6: rep += bias
// ---------------------------------------------------------------------------
__global__ void k_bias(float* __restrict__ rep, const float* __restrict__ bias) {
    int idx = blockIdx.x * blockDim.x + threadIdx.x;
    if (idx >= B * N * DEN) return;
    rep[idx] += bias[idx & (DEN - 1)];
}

// ---------------------------------------------------------------------------
// K7: row_ptr[b][i] = lower_bound(src_b, i)  (src is sorted, covers [0,NREG))
// ---------------------------------------------------------------------------
__global__ void k_rowptr(const long long* __restrict__ ei, int* __restrict__ row_ptr) {
    int idx = blockIdx.x * blockDim.x + threadIdx.x;
    if (idx >= B * (NREG + 1)) return;
    const int b = idx / (NREG + 1);
    const long long i = idx - b * (NREG + 1);
    const long long* src = ei + (long long)b * 2 * E;
    int lo = 0, hi = E;
    while (lo < hi) {
        const int mid = (lo + hi) >> 1;
        if (src[mid] < i) lo = mid + 1; else hi = mid;
    }
    row_ptr[idx] = lo;
}

// ---------------------------------------------------------------------------
// K8: regulon pooling — edge (s,d) contributes rep[d] to pool1[s] iff
// multiplicity of (s,d) within s's contiguous block is exactly 1.
// ---------------------------------------------------------------------------
__global__ void k_pool(const long long* __restrict__ ei,
                       const int* __restrict__ row_ptr,
                       const float* __restrict__ rep,
                       float* __restrict__ pool1) {
    int idx = blockIdx.x * blockDim.x + threadIdx.x;
    if (idx >= B * E) return;
    const int b = idx / E, e = idx - b * E;
    const long long* src = ei + (long long)b * 2 * E;
    const long long* dst = src + E;
    const int s = (int)src[e];
    const long long d = dst[e];
    const int lo = row_ptr[b * (NREG + 1) + s];
    const int hi = row_ptr[b * (NREG + 1) + s + 1];
    int cnt = 0;
    for (int j = lo; j < hi; ++j) cnt += (dst[j] == d) ? 1 : 0;
    if (cnt == 1) {
        const float* rd = rep + (long long)(b * N + (int)d) * DEN;
        float* p = pool1 + (long long)(b * NREG + s) * DEN;
        #pragma unroll
        for (int c = 0; c < DEN; ++c) atomicAdd(&p[c], rd[c]);
    }
}

// ---------------------------------------------------------------------------
// K9: out = rep[:NREG] * pool1
// ---------------------------------------------------------------------------
__global__ void k_final(const float* __restrict__ rep,
                        const float* __restrict__ pool1,
                        float* __restrict__ out) {
    int idx = blockIdx.x * blockDim.x + threadIdx.x;
    if (idx >= B * NREG * DEN) return;
    const int b = idx / (NREG * DEN);
    const int r = (idx / DEN) % NREG;
    const int c = idx & (DEN - 1);
    out[idx] = rep[((long long)b * N + r) * DEN + c] * pool1[idx];
}

// ---------------------------------------------------------------------------
extern "C" void kernel_launch(void* const* d_in, const int* in_sizes, int n_in,
                              void* d_out, int out_size, void* d_ws, size_t ws_size,
                              hipStream_t stream) {
    const float*     x        = (const float*)d_in[0];      // [B,N,DIN]
    const long long* ei       = (const long long*)d_in[1];  // [B,2,E] int64
    const float*     ea       = (const float*)d_in[2];      // [B,E,1]
    const float*     Wm       = (const float*)d_in[3];      // [DIN,DEN]
    const float*     att_src  = (const float*)d_in[4];      // [DEN]
    const float*     att_dst  = (const float*)d_in[5];      // [DEN]
    const float*     lin_edge = (const float*)d_in[6];      // [1,DEN]
    const float*     att_edge = (const float*)d_in[7];      // [DEN]
    const float*     bias     = (const float*)d_in[8];      // [DEN]
    float*           out      = (float*)d_out;              // [B,NREG,DEN]

    // Workspace carve-up (all L2-resident, ~9 MB total)
    float* h       = (float*)d_ws;                // B*N*DEN
    float* a_s     = h + (size_t)B * N * DEN;     // B*N
    float* a_d     = a_s + (size_t)B * N;         // B*N
    float* m       = a_d + (size_t)B * N;         // B*N
    float* den     = m + (size_t)B * N;           // B*N
    float* alpha   = den + (size_t)B * N;         // B*E (alpha, then ex, in place)
    float* rep     = alpha + (size_t)B * E;       // B*N*DEN
    float* pool1   = rep + (size_t)B * N * DEN;   // B*NREG*DEN
    float* ce      = pool1 + (size_t)B * NREG * DEN; // 1
    int*   row_ptr = (int*)(ce + 1);              // B*(NREG+1)

    const int T = 256;
    auto blocks = [](long long n, int t) { return (int)((n + t - 1) / t); };

    // K0: init (covers the largest zeroed region: B*N*DEN)
    k_init<<<blocks((long long)B * N * DEN, T), T, 0, stream>>>(
        m, den, rep, pool1, ce, lin_edge, att_edge);

    // K1: WMMA GEMM, 2500 tiles, 8 waves / block
    {
        const int tiles = (B * N) / 16;
        const int wavesPerBlock = T / 32;
        k_gemm_wmma<<<blocks(tiles, wavesPerBlock), T, 0, stream>>>(x, Wm, h);
    }

    // K2: node projections
    k_node_proj<<<blocks((long long)B * N, T), T, 0, stream>>>(h, att_src, att_dst, a_s, a_d);

    // K3: alpha + segment max
    k_alpha_max<<<blocks((long long)B * E, T), T, 0, stream>>>(ei, ea, a_s, a_d, ce, alpha, m);

    // K4: exp + segment sum
    k_exp_sum<<<blocks((long long)B * E, T), T, 0, stream>>>(ei, m, alpha, den);

    // K5: attention-weighted scatter
    k_scatter<<<blocks((long long)B * E, T), T, 0, stream>>>(ei, alpha, den, h, rep);

    // K6: bias
    k_bias<<<blocks((long long)B * N * DEN, T), T, 0, stream>>>(rep, bias);

    // K7: CSR row pointers for the sorted regulator array
    k_rowptr<<<blocks((long long)B * (NREG + 1), T), T, 0, stream>>>(ei, row_ptr);

    // K8: regulon pooling (multiplicity == 1 edges)
    k_pool<<<blocks((long long)B * E, T), T, 0, stream>>>(ei, row_ptr, rep, pool1);

    // K9: final elementwise product
    k_final<<<blocks((long long)B * NREG * DEN, T), T, 0, stream>>>(rep, pool1, out);

    (void)in_sizes; (void)n_in; (void)out_size; (void)ws_size;
}
